// QLSTM__gen361_65481071400996
// MI455X (gfx1250) — matrix operand order repaired and checked by reading, not compiled, not run on hardware
//
#include <hip/hip_runtime.h>

// ---------------- problem constants ----------------
#define S_LEN 512
#define BATCH 64
#define DIM   512
#define HID   512
#define DH    1024                 // D + H
#define NWG   16                   // persistent workgroups
#define TPB   256                  // 8 waves (wave32)
#define GATE_STRIDE (HID * DH)     // 524288 bf16 elems per gate matrix
#define BH    (BATCH * HID)        // 32768
#define BD    (BATCH * DIM)        // 32768
#define SBH   (S_LEN * BH)

// ---------------- clang ext-vector types (POD, union-safe) ----------------
typedef __attribute__((ext_vector_type(16))) __bf16    bf16x16;
typedef __attribute__((ext_vector_type(8)))  float     floatx8;
typedef __attribute__((ext_vector_type(4)))  unsigned  uintx4;
typedef __attribute__((ext_vector_type(2)))  unsigned  uintx2;
typedef __attribute__((ext_vector_type(4)))  float     fltx4;
typedef __attribute__((ext_vector_type(8)))  int       intx8;
typedef __attribute__((ext_vector_type(4)))  int       intx4;

union FragU { uintx4 q[2]; bf16x16 v; };

// ---------------- helpers ----------------
__device__ inline unsigned short f2bf(float f) {
  unsigned u = __float_as_uint(f);
  u += 0x7FFFu + ((u >> 16) & 1u);          // round-to-nearest-even
  return (unsigned short)(u >> 16);
}
__device__ inline float fsigmoid(float x) {
  return __frcp_rn(1.0f + __expf(-x));
}
__device__ inline float ftanh(float x) {
  x = fminf(15.0f, fmaxf(-15.0f, x));
  float e = __expf(-2.0f * x);
  return (1.0f - e) * __frcp_rn(1.0f + e);
}

// ---------------- TDM: DMA a 64x512 bf16 tile (contiguous) global -> LDS -------
// Descriptor per CDNA5 ISA 08_async_tensor §8.3/8.4: 2D tensor, data_size=2B,
// tensor_dim0=512 (stride 512), tensor_dim1=64, tile 512x64; groups 2/3 dim=1.
__device__ inline void tdm_load_tile_64x512(unsigned lds_byte_off,
                                            const unsigned short* gsrc) {
  unsigned long long ga = (unsigned long long)(uintptr_t)gsrc;
  uintx4 g0;
  g0.x = 1u;                                           // count=1, user mode
  g0.y = lds_byte_off;                                 // lds_addr [63:32]
  g0.z = (unsigned)(ga & 0xFFFFFFFFull);               // global_addr lo
  g0.w = (unsigned)((ga >> 32) & 0x1FFFFFFull)         // global_addr hi (57b)
         | (2u << 30);                                 // type = 2 ("image")
  intx8 g1;
  g1[0] = 0x00010000;        // data_size=1 (2 bytes) at [17:16]
  g1[1] = (int)0x02000000;   // tensor_dim0 = 512  -> bits[79:48] low16 << 16
  g1[2] = (int)0x00400000;   // tensor_dim1 = 64   -> bits[111:80] low16 << 16
  g1[3] = (int)0x02000000;   // tile_dim0  = 512   -> bits[127:112]
  g1[4] = (int)0x00010040;   // tile_dim1 = 64 | tile_dim2 = 1 << 16
  g1[5] = 512;               // tensor_dim0_stride low32
  g1[6] = (int)0x02000000;   // tensor_dim1_stride = 512 -> low16 << 16
  g1[7] = 0;
  intx4 g2;                  // tensor_dim2=1, tensor_dim3=1, stride2=0, tile_dim3=1
  g2[0] = 1; g2[1] = 1; g2[2] = 0; g2[3] = 0x00010000;
  intx4 g3;                  // stride3=0, tensor_dim4=1, tile_dim4=1
  g3[0] = 0; g3[1] = 0x00010000; g3[2] = 0x00010000; g3[3] = 0;
#if defined(__clang_major__) && (__clang_major__ >= 23)
  intx8 g4 = {0, 0, 0, 0, 0, 0, 0, 0};
  __builtin_amdgcn_tensor_load_to_lds(g0, g1, g2, g3, g4, 0);
#else
  __builtin_amdgcn_tensor_load_to_lds(g0, g1, g2, g3, 0);
#endif
}

// ---------------- prep: f32 weights -> bf16 [4][H][DH] ----------------
__global__ __launch_bounds__(256) void wconv_kernel(
    const float* __restrict__ Wf, const float* __restrict__ Wi,
    const float* __restrict__ Wg, const float* __restrict__ Wo,
    unsigned short* __restrict__ dst) {
  int i = blockIdx.x * 256 + threadIdx.x;          // 8192*256 == 4*GATE_STRIDE
  int gate = i >> 19;
  int idx  = i & (GATE_STRIDE - 1);
  const float* src = (gate == 0) ? Wf : (gate == 1) ? Wi : (gate == 2) ? Wg : Wo;
  dst[i] = f2bf(src[idx]);
}

// ---------------- prep: f32 inputs (S,B,D) -> bf16 (parallel, off serial path) --
__global__ __launch_bounds__(256) void xconv_kernel(const float* __restrict__ x,
                                                    unsigned short* __restrict__ xbf) {
  int i = blockIdx.x * 256 + threadIdx.x;          // 16384*256 float4 == S*B*D/4
  fltx4 v = reinterpret_cast<const fltx4*>(x)[i];
  uintx2 p;
  p.x = (unsigned)f2bf(v.x) | ((unsigned)f2bf(v.y) << 16);
  p.y = (unsigned)f2bf(v.z) | ((unsigned)f2bf(v.w) << 16);
  *reinterpret_cast<uintx2*>(&xbf[(size_t)i * 4]) = p;
}

// ---------------- prep: zero h double-buffer + barrier words ----------------
__global__ __launch_bounds__(256) void init_kernel(unsigned short* __restrict__ hbuf,
                                                   unsigned* __restrict__ bar) {
  int i = blockIdx.x * 256 + threadIdx.x;          // 256*256 == 2*BH
  hbuf[i] = 0;
  if (i < 2) bar[i] = 0u;
}

// ---------------- persistent LSTM kernel ----------------
__global__ __launch_bounds__(TPB) void lstm_persist_kernel(
    const unsigned short* __restrict__ xbf,        // (S, B, D) bf16
    const float* __restrict__ bF, const float* __restrict__ bI,
    const float* __restrict__ bG, const float* __restrict__ bO,
    const unsigned short* __restrict__ wbf,        // [4][HID][DH] bf16
    unsigned short* __restrict__ hbuf,             // [2][B][HID] bf16 double buffer
    unsigned* __restrict__ bar,                    // [counter, generation]
    float* __restrict__ out)                       // outputs (S,B,H) ++ hx ++ cx
{
  // dynamic LDS, 192 KB: [0,64K) h tile, [64K,128K) x tile buf0, [128K,192K) buf1
  extern __shared__ unsigned short smem_dyn[];
  unsigned short* hL = smem_dyn;
  const unsigned ldsbase = (unsigned)(uintptr_t)(void*)smem_dyn; // LDS byte offset

  const int tid   = threadIdx.x;
  const int lane  = tid & 31;
  const int wid   = tid >> 5;
  const int mbase = (wid & 3) * 16;                     // batch-tile base
  const int ubase = blockIdx.x * 32 + (wid >> 2) * 16;  // hidden-tile base
  const int l16   = lane & 15;
  const int lhi   = lane >> 4;                          // 0 / 1
  const int ucol  = ubase + l16;                        // hidden col == weight row
  const int arow  = mbase + l16;                        // batch row for A frags

  const float biasF = bF[ucol], biasI = bI[ucol], biasG = bG[ucol], biasO = bO[ucol];

  const unsigned short* wFp = wbf + 0 * GATE_STRIDE + (size_t)ucol * DH;
  const unsigned short* wIp = wbf + 1 * GATE_STRIDE + (size_t)ucol * DH;
  const unsigned short* wGp = wbf + 2 * GATE_STRIDE + (size_t)ucol * DH;
  const unsigned short* wOp = wbf + 3 * GATE_STRIDE + (size_t)ucol * DH;

  floatx8 creg = {};                                    // c state lives in VGPRs

  // prefetch x(0) into x LDS buffer 0 via Tensor Data Mover
  if (wid == 0) {
    tdm_load_tile_64x512(ldsbase + 65536u, xbf);
  }

  for (int s = 0; s < S_LEN; ++s) {
    floatx8 aF = {}, aI = {}, aG = {}, aO = {};

    // ---- TDM stage: h(s) now; prefetch x(s+1) behind the compute ----
    if (wid == 0) {
      tdm_load_tile_64x512(ldsbase, hbuf + (size_t)(s & 1) * BH);
      if (s < S_LEN - 1) {
        tdm_load_tile_64x512(ldsbase + 65536u + (unsigned)((s + 1) & 1) * 65536u,
                             xbf + (size_t)(s + 1) * BD);
        // in-order completion: tensorcnt<=1 => x(s) and h(s) are done,
        // x(s+1) stays in flight through this step's WMMA work
        __builtin_amdgcn_s_wait_tensorcnt(1);
      } else {
        __builtin_amdgcn_s_wait_tensorcnt(0);
      }
    }
    __syncthreads();

    const unsigned short* xrp = smem_dyn + 32768 + (s & 1) * 32768 + arow * 512;
    const unsigned short* hrp = hL + arow * 512;

#define K_CHUNK(AP_BASE, KB_W, KL_S)                                                 \
    {                                                                                \
      FragU a;                                                                       \
      const uintx4* ap = reinterpret_cast<const uintx4*>((AP_BASE) + (KL_S));        \
      a.q[0] = ap[0]; a.q[1] = ap[2];          /* chunks K+0..7 and K+16..23 */      \
      FragU fb, ib, gb, ob;                                                          \
      { const uintx4* p = reinterpret_cast<const uintx4*>(wFp + (KB_W));             \
        fb.q[0] = p[0]; fb.q[1] = p[2]; }                                            \
      { const uintx4* p = reinterpret_cast<const uintx4*>(wIp + (KB_W));             \
        ib.q[0] = p[0]; ib.q[1] = p[2]; }                                            \
      { const uintx4* p = reinterpret_cast<const uintx4*>(wGp + (KB_W));             \
        gb.q[0] = p[0]; gb.q[1] = p[2]; }                                            \
      { const uintx4* p = reinterpret_cast<const uintx4*>(wOp + (KB_W));             \
        ob.q[0] = p[0]; ob.q[1] = p[2]; }                                            \
      aF = __builtin_amdgcn_wmma_f32_16x16x32_bf16(false, a.v, false, fb.v,          \
                                                   (short)0, aF, false, false);      \
      aI = __builtin_amdgcn_wmma_f32_16x16x32_bf16(false, a.v, false, ib.v,          \
                                                   (short)0, aI, false, false);      \
      aG = __builtin_amdgcn_wmma_f32_16x16x32_bf16(false, a.v, false, gb.v,          \
                                                   (short)0, aG, false, false);      \
      aO = __builtin_amdgcn_wmma_f32_16x16x32_bf16(false, a.v, false, ob.v,          \
                                                   (short)0, aO, false, false);      \
    }

    // ---- K = 0..511 : the x half of comb (prefetched last step) ----
    for (int kc = 0; kc < 16; ++kc) {
      const int klocal = kc * 32 + lhi * 8;     // LDS: per-lane A chunk start
      const int kbw    = kc * 32 + lhi * 8;     // weight row: K+0..7 and K+16..23
      K_CHUNK(xrp, kbw, klocal)
    }
    // ---- K = 512..1023 : the h half of comb ----
    for (int kc = 0; kc < 16; ++kc) {
      const int klocal = kc * 32 + lhi * 8;
      const int kbw    = 512 + kc * 32 + lhi * 8;
      K_CHUNK(hrp, kbw, klocal)
    }
#undef K_CHUNK

    // ---- gate combine (all 4 gates in-lane), c in registers ----
    unsigned short* hn   = hbuf + (size_t)((s + 1) & 1) * BH;
    float*          orow = out + (size_t)s * BH;
#pragma unroll
    for (int i = 0; i < 8; ++i) {
      const int b = mbase + i + lhi * 8;            // D-matrix row mapping (ISA 7.12.2)
      float fg = fsigmoid(aF[i] + biasF);
      float ig = fsigmoid(aI[i] + biasI);
      float gg = ftanh   (aG[i] + biasG);
      float og = fsigmoid(aO[i] + biasO);
      float c  = fg * creg[i] + ig * gg;
      creg[i]  = c;
      float h  = og * ftanh(c);
      orow[b * HID + ucol] = h;
      hn[b * HID + ucol]   = f2bf(h);
      if (s == S_LEN - 1) {
        out[SBH + b * HID + ucol]      = h;         // hx
        out[SBH + BH + b * HID + ucol] = c;         // cx
      }
    }

    // ---- device-wide barrier: one per timestep ----
    __threadfence();
    __syncthreads();
    if (tid == 0) {
      volatile unsigned* vgen = bar + 1;
      unsigned g = *vgen;
      if (atomicAdd(bar, 1u) == NWG - 1) {
        bar[0] = 0u;
        __threadfence();
        atomicAdd((unsigned*)(bar + 1), 1u);        // release
      } else {
        while (*vgen == g) { __builtin_amdgcn_s_sleep(2); }
      }
      __threadfence();
    }
    __syncthreads();
  }
}

// ---------------- host entry ----------------
extern "C" void kernel_launch(void* const* d_in, const int* in_sizes, int n_in,
                              void* d_out, int out_size, void* d_ws, size_t ws_size,
                              hipStream_t stream) {
  const float* x  = (const float*)d_in[0];
  const float* Wf = (const float*)d_in[1];
  const float* bf = (const float*)d_in[2];
  const float* Wi = (const float*)d_in[3];
  const float* bi = (const float*)d_in[4];
  const float* Wg = (const float*)d_in[5];
  const float* bg = (const float*)d_in[6];
  const float* Wo = (const float*)d_in[7];
  const float* bo = (const float*)d_in[8];

  // ws layout: bf16 weights [4][H][DH] | bf16 x (S,B,D) | bf16 h x2 | barrier
  unsigned short* wbf  = (unsigned short*)d_ws;                 // 4 MiB
  unsigned short* xbf  = wbf + 4 * GATE_STRIDE;                 // 32 MiB
  unsigned short* hbuf = xbf + (size_t)S_LEN * BD;              // 128 KiB
  unsigned*       bar  = (unsigned*)(hbuf + 2 * BH);            // 8 B

  wconv_kernel<<<8192, 256, 0, stream>>>(Wf, Wi, Wg, Wo, wbf);
  xconv_kernel<<<16384, 256, 0, stream>>>(x, xbf);
  init_kernel<<<256, 256, 0, stream>>>(hbuf, bar);
  lstm_persist_kernel<<<NWG, TPB, 3 * 65536, stream>>>(
      xbf, bf, bi, bg, bo, wbf, hbuf, bar, (float*)d_out);
}
